// NWGformer_60198261620975
// MI455X (gfx1250) — compile-verified, matrix operand order checked by reading, not run on hardware
//
#include <hip/hip_runtime.h>
#include <hip/hip_bf16.h>
#include <math.h>

typedef __attribute__((ext_vector_type(16))) _Float16 v16h;
typedef __attribute__((ext_vector_type(8)))  float    v8f;

constexpr int DD = 512;   // feature dim
constexpr float LN_EPS = 1e-5f;
constexpr float NORM_EPS = 1e-10f;

// ---------------------------------------------------------------------------
// CDNA5 async global->LDS copy path (ASYNCcnt). Guarded: falls back to plain
// LDS staging if the builtins aren't declared by this toolchain.
// Builtin signature (from hipcc diagnostic): first param is
//   int __attribute__((vector_size(16))) addrspace(1)*   (global int4*)
// so pass typed int4 pointers in AS(1)/AS(3).
// ---------------------------------------------------------------------------
#if defined(__has_builtin)
#if __has_builtin(__builtin_amdgcn_global_load_async_to_lds_b128)
#define NWG_ASYNC 1
#endif
#endif

#if defined(__has_builtin)
#if __has_builtin(__builtin_amdgcn_s_wait_asynccnt)
#define NWG_WAIT_ASYNC() __builtin_amdgcn_s_wait_asynccnt(0)
#else
#define NWG_WAIT_ASYNC() asm volatile("s_wait_asynccnt 0x0" ::: "memory")
#endif
#else
#define NWG_WAIT_ASYNC() asm volatile("s_wait_asynccnt 0x0" ::: "memory")
#endif

#ifdef NWG_ASYNC
typedef int nwg_v4i __attribute__((vector_size(16)));
typedef __attribute__((address_space(1))) nwg_v4i g_v4i;   // global int4
typedef __attribute__((address_space(3))) nwg_v4i l_v4i;   // LDS int4
__device__ __forceinline__ void async_cp16(const _Float16* g, _Float16* l) {
  // copy 16 bytes (8 halves) global -> LDS, tracked by ASYNCcnt
  __builtin_amdgcn_global_load_async_to_lds_b128((g_v4i*)g, (l_v4i*)l, 0, 0);
}
#endif

// ---------------------------------------------------------------------------
// WMMA fragment helpers (CDNA5 16x16x32 f16 layouts, wave32)
// A frag (16x32, MxK): lane m = lane&15; vgpr v holds K pair
//   k0 = (v<4 ? 2v : 16+2(v-4)) + 8*(lane>=16)
// B frag staged transposed in LDS as [n][k] so same loader applies (n = lane&15)
// ---------------------------------------------------------------------------
__device__ __forceinline__ v16h load_frag(const _Float16* __restrict__ s,
                                          int row, int stride, int hi) {
  v16h f;
#pragma unroll
  for (int v = 0; v < 8; ++v) {
    int k = ((v < 4) ? (v * 2) : (16 + (v - 4) * 2)) + hi * 8;
    f[2 * v]     = s[row * stride + k];
    f[2 * v + 1] = s[row * stride + k + 1];
  }
  return f;
}

__device__ __forceinline__ v16h scale_frag(v16h a, float fs) {
  _Float16 h = (_Float16)fs;
#pragma unroll
  for (int i = 0; i < 16; ++i) a[i] = a[i] * h;
  return a;
}

__device__ __forceinline__ float wave_sum(float v) {
#pragma unroll
  for (int o = 16; o > 0; o >>= 1) v += __shfl_xor(v, o, 32);
  return v;
}

// ---------------------------------------------------------------------------
// K0: init scratch (KV accumulators, colsums, deg=1 for self loop, H min/max)
// ---------------------------------------------------------------------------
__global__ void nwg_k0_init(float* KVc, float* KVs, float* colc, float* cols_,
                            float* deg, unsigned* hmax, unsigned* hmin, int Nn) {
  int i = blockIdx.x * 256 + threadIdx.x;
  int stride = gridDim.x * 256;
  for (int j = i; j < DD * DD; j += stride) { KVc[j] = 0.f; KVs[j] = 0.f; }
  for (int j = i; j < DD; j += stride) { colc[j] = 0.f; cols_[j] = 0.f; }
  for (int j = i; j < Nn; j += stride) deg[j] = 1.0f;   // self loop
  if (i == 0) { *hmax = 0u; *hmin = 0x7F800000u; }
}

// ---------------------------------------------------------------------------
// Kpre: x (f32) -> xh (f16) once, so GEMM staging can be a verbatim async copy
// ---------------------------------------------------------------------------
__global__ void nwg_kpre_cvt(const float* __restrict__ x, _Float16* __restrict__ xh,
                             long long n) {
  long long i = (long long)blockIdx.x * 256 + threadIdx.x;
  long long stride = (long long)gridDim.x * 256;
  for (long long j = i; j < n; j += stride) xh[j] = (_Float16)x[j];
}

// ---------------------------------------------------------------------------
// K1: fused triple GEMM.  Q=relu(xWq+bq)->f16, V=xWv+bv->f32+f16, XW=xWg->f32
// Block: 256 thr (8 waves), tile 128 rows x 64 cols per output, K-step 32.
// A tile staged with GLOBAL_LOAD_ASYNC_TO_LDS_B128 (verbatim f16 copy).
// ---------------------------------------------------------------------------
__global__ __launch_bounds__(256)
void nwg_k1_gemm3(const _Float16* __restrict__ xh,
                  const float* __restrict__ Wq, const float* __restrict__ bq,
                  const float* __restrict__ Wv, const float* __restrict__ bv,
                  const float* __restrict__ Wg,
                  _Float16* __restrict__ Qh, float* __restrict__ Vf,
                  _Float16* __restrict__ Vh, float* __restrict__ XWf, int Nn) {
  __shared__ _Float16 sA[128 * 40];
  __shared__ _Float16 sB[3][64 * 40];
  const int tid = threadIdx.x, lane = tid & 31, wave = tid >> 5;
  const int rowBase = blockIdx.x * 128;
  const int colBase = blockIdx.y * 64;

  v8f accQ[4], accV[4], accG[4];
  v8f z = {0.f, 0.f, 0.f, 0.f, 0.f, 0.f, 0.f, 0.f};
#pragma unroll
  for (int t = 0; t < 4; ++t) { accQ[t] = z; accV[t] = z; accG[t] = z; }

  for (int k0 = 0; k0 < DD; k0 += 32) {
    __syncthreads();
#ifdef NWG_ASYNC
    // stage x tile 128x32 halves as 512 16-byte async chunks (2 per thread)
    for (int i = tid; i < 512; i += 256) {
      int r = i >> 2, c = (i & 3) * 8;                  // c in halves
      int gr = min(rowBase + r, Nn - 1);                // clamp: pad rows masked later
      async_cp16(&xh[(size_t)gr * DD + k0 + c], &sA[r * 40 + c]);
    }
#else
    for (int i = tid; i < 128 * 32; i += 256) {
      int r = i >> 5, c = i & 31;
      int gr = min(rowBase + r, Nn - 1);
      sA[r * 40 + c] = xh[(size_t)gr * DD + k0 + c];
    }
#endif
    // stage weight tiles 32x64 (f32->f16), transposed to [n][k]
    for (int i = tid; i < 32 * 64; i += 256) {
      int kk = i >> 6, nn = i & 63;
      int gidx = (k0 + kk) * DD + colBase + nn;
      sB[0][nn * 40 + kk] = (_Float16)Wq[gidx];
      sB[1][nn * 40 + kk] = (_Float16)Wv[gidx];
      sB[2][nn * 40 + kk] = (_Float16)Wg[gidx];
    }
#ifdef NWG_ASYNC
    NWG_WAIT_ASYNC();
#endif
    __syncthreads();
    int hi = lane >> 4;
    v16h a = load_frag(sA, wave * 16 + (lane & 15), 40, hi);
#pragma unroll
    for (int t = 0; t < 4; ++t) {
      int n = t * 16 + (lane & 15);
      v16h b0 = load_frag(sB[0], n, 40, hi);
      accQ[t] = __builtin_amdgcn_wmma_f32_16x16x32_f16(false, a, false, b0,
                                                       (short)0, accQ[t], false, false);
      v16h b1 = load_frag(sB[1], n, 40, hi);
      accV[t] = __builtin_amdgcn_wmma_f32_16x16x32_f16(false, a, false, b1,
                                                       (short)0, accV[t], false, false);
      v16h b2 = load_frag(sB[2], n, 40, hi);
      accG[t] = __builtin_amdgcn_wmma_f32_16x16x32_f16(false, a, false, b2,
                                                       (short)0, accG[t], false, false);
    }
  }

  int hi = lane >> 4;
#pragma unroll
  for (int t = 0; t < 4; ++t) {
    int col = colBase + t * 16 + (lane & 15);
#pragma unroll
    for (int r = 0; r < 8; ++r) {
      int row = rowBase + wave * 16 + hi * 8 + r;
      if (row < Nn) {
        size_t idx = (size_t)row * DD + col;
        float q = fmaxf(accQ[t][r] + bq[col], 0.f);
        Qh[idx] = (_Float16)q;
        float v = accV[t][r] + bv[col];
        Vf[idx] = v;
        Vh[idx] = (_Float16)v;
        XWf[idx] = accG[t][r];
      }
    }
  }
}

// ---------------------------------------------------------------------------
// K_deg / K_dinv: GCN degree with self loops -> d^-1/2
// ---------------------------------------------------------------------------
__global__ void nwg_k_deg(const int* __restrict__ ei, float* deg, int E_) {
  int e = blockIdx.x * 256 + threadIdx.x;
  if (e < E_) atomicAdd(&deg[ei[E_ + e]], 1.0f);   // dst row of edge_index
}
__global__ void nwg_k_dinv(const float* deg, float* dinv, int Nn) {
  int i = blockIdx.x * 256 + threadIdx.x;
  if (i < Nn) dinv[i] = rsqrtf(fmaxf(deg[i], 1e-12f));
}

// ---------------------------------------------------------------------------
// K2: per-row parameter-free LN + relu on Qh (in place, f16), rowsum^2 -> H,
//     atomic min/max of H (H>=0 so uint-bit compare is monotonic)
// ---------------------------------------------------------------------------
__global__ __launch_bounds__(256)
void nwg_k2_ln(_Float16* Qh, float* H, unsigned* hmax, unsigned* hmin, int Nn) {
  int row = blockIdx.x * 8 + (threadIdx.x >> 5);
  int lane = threadIdx.x & 31;
  if (row >= Nn) return;
  size_t base = (size_t)row * DD + lane * 16;
  float v[16]; float s = 0.f;
#pragma unroll
  for (int i = 0; i < 16; ++i) { v[i] = (float)Qh[base + i]; s += v[i]; }
  float mu = wave_sum(s) * (1.0f / DD);
  float var = 0.f;
#pragma unroll
  for (int i = 0; i < 16; ++i) { float d = v[i] - mu; var += d * d; }
  var = wave_sum(var) * (1.0f / DD);
  float inv = rsqrtf(var + LN_EPS);
  float rs = 0.f;
#pragma unroll
  for (int i = 0; i < 16; ++i) {
    float q = fmaxf((v[i] - mu) * inv, 0.f);
    Qh[base + i] = (_Float16)q;
    rs += q;
  }
  rs = wave_sum(rs);
  if (lane == 0) {
    float h = rs * rs;
    H[row] = h;
    atomicMax(hmax, __float_as_uint(h));
    atomicMin(hmin, __float_as_uint(h));
  }
}

// ---------------------------------------------------------------------------
// K3: per-node cos/sin of scaled H
// ---------------------------------------------------------------------------
__global__ void nwg_k3_trig(const float* H, const unsigned* hmax,
                            const unsigned* hmin, float* cv, float* sv, int Nn) {
  float m = (__uint_as_float(*hmax) - __uint_as_float(*hmin)) * 2.0f; // M_SCALE
  float scale = 3.14159265358979f / (2.0f * m);
  int i = blockIdx.x * 256 + threadIdx.x;
  if (i < Nn) { float h = scale * H[i]; cv[i] = cosf(h); sv[i] = sinf(h); }
}

// ---------------------------------------------------------------------------
// K4: column sums of cosQ / sinQ
// ---------------------------------------------------------------------------
__global__ __launch_bounds__(512)
void nwg_k4_colsum(const _Float16* Qh, const float* cv, const float* sv,
                   float* colc, float* cols_, int Nn) {
  int d = threadIdx.x;
  int r0 = blockIdx.x * 256;
  int r1 = min(r0 + 256, Nn);
  float ac = 0.f, as = 0.f;
  for (int r = r0; r < r1; ++r) {
    float q = (float)Qh[(size_t)r * DD + d];
    ac += q * cv[r]; as += q * sv[r];
  }
  atomicAdd(&colc[d], ac);
  atomicAdd(&cols_[d], as);
}

// ---------------------------------------------------------------------------
// K5: Gram GEMMs KVc = cosQ^T @ V, KVs = sinQ^T @ V  (split-K over nodes,
// f32 atomic accumulate). Per-K scaling by c[node]/s[node] at staging time.
// ---------------------------------------------------------------------------
__global__ __launch_bounds__(256)
void nwg_k5_gram(const _Float16* __restrict__ Qh, const _Float16* __restrict__ Vh,
                 const float* __restrict__ cv, const float* __restrict__ sv,
                 float* KVc, float* KVs, int Nn) {
  __shared__ _Float16 sAc[64 * 40];
  __shared__ _Float16 sAs[64 * 40];
  __shared__ _Float16 sB[64 * 40];
  const int tid = threadIdx.x, lane = tid & 31, wave = tid >> 5;
  const int d1base = blockIdx.x * 64;
  const int d2base = blockIdx.y * 64;
  int chunk = ((Nn + gridDim.z - 1) / gridDim.z + 31) & ~31;
  int kstart = blockIdx.z * chunk;
  int kend = min(kstart + chunk, Nn);

  const int mtile = wave >> 1;        // 0..3
  const int nhalf = wave & 1;         // 0..1

  v8f z = {0.f, 0.f, 0.f, 0.f, 0.f, 0.f, 0.f, 0.f};
  v8f accC[2] = {z, z}, accS[2] = {z, z};

  for (int k0 = kstart; k0 < kend; k0 += 32) {
    __syncthreads();
    for (int i = tid; i < 32 * 64; i += 256) {
      int kk = i >> 6, mm = i & 63;
      int node = k0 + kk;
      float q = 0.f, vb = 0.f, cn = 0.f, sn = 0.f;
      if (node < Nn) {
        q = (float)Qh[(size_t)node * DD + d1base + mm];
        vb = (float)Vh[(size_t)node * DD + d2base + mm];
        cn = cv[node]; sn = sv[node];
      }
      sAc[mm * 40 + kk] = (_Float16)(q * cn);
      sAs[mm * 40 + kk] = (_Float16)(q * sn);
      sB[mm * 40 + kk] = (_Float16)vb;
    }
    __syncthreads();
    int hi = lane >> 4;
    v16h ac = load_frag(sAc, mtile * 16 + (lane & 15), 40, hi);
    v16h as = load_frag(sAs, mtile * 16 + (lane & 15), 40, hi);
#pragma unroll
    for (int nt = 0; nt < 2; ++nt) {
      int n = nhalf * 32 + nt * 16 + (lane & 15);
      v16h b = load_frag(sB, n, 40, hi);
      accC[nt] = __builtin_amdgcn_wmma_f32_16x16x32_f16(false, ac, false, b,
                                                        (short)0, accC[nt], false, false);
      accS[nt] = __builtin_amdgcn_wmma_f32_16x16x32_f16(false, as, false, b,
                                                        (short)0, accS[nt], false, false);
    }
  }

  int hi = lane >> 4;
#pragma unroll
  for (int nt = 0; nt < 2; ++nt) {
    int col = d2base + nhalf * 32 + nt * 16 + (lane & 15);
#pragma unroll
    for (int r = 0; r < 8; ++r) {
      int d1 = d1base + mtile * 16 + hi * 8 + r;
      atomicAdd(&KVc[d1 * DD + col], accC[nt][r]);
      atomicAdd(&KVs[d1 * DD + col], accS[nt][r]);
    }
  }
}

// ---------------------------------------------------------------------------
// K6: f32 KV matrices -> f16 for the back-projection GEMM
// ---------------------------------------------------------------------------
__global__ void nwg_k6_cvt(const float* KVc, const float* KVs,
                           _Float16* KVch, _Float16* KVsh) {
  int i = blockIdx.x * 256 + threadIdx.x;
  if (i < DD * DD) { KVch[i] = (_Float16)KVc[i]; KVsh[i] = (_Float16)KVs[i]; }
}

// ---------------------------------------------------------------------------
// K7b: per-row attention normalizer
// ---------------------------------------------------------------------------
__global__ __launch_bounds__(256)
void nwg_k7b_norm(const _Float16* Qh, const float* colc, const float* cols_,
                  const float* cv, const float* sv, float* normv, int Nn) {
  int row = blockIdx.x * 8 + (threadIdx.x >> 5);
  int lane = threadIdx.x & 31;
  if (row >= Nn) return;
  float dc = 0.f, ds = 0.f;
#pragma unroll
  for (int i = 0; i < 16; ++i) {
    int d = lane * 16 + i;
    float q = (float)Qh[(size_t)row * DD + d];
    dc += q * colc[d]; ds += q * cols_[d];
  }
  dc = wave_sum(dc); ds = wave_sum(ds);
  if (lane == 0) normv[row] = cv[row] * dc + sv[row] * ds + NORM_EPS;
}

// ---------------------------------------------------------------------------
// K7: back-projection GEMM + assembly:
//   d_out = (cosQ@KVc + sinQ@KVs)/norm + V + bg + dinv^2 * XW   (self loop)
// A tile staged async; row scalars c_i/s_i applied to the A frag in registers.
// ---------------------------------------------------------------------------
__global__ __launch_bounds__(256)
void nwg_k7_out(const _Float16* __restrict__ Qh,
                const _Float16* __restrict__ KVch, const _Float16* __restrict__ KVsh,
                const float* __restrict__ cv, const float* __restrict__ sv,
                const float* __restrict__ normv, const float* __restrict__ Vf,
                const float* __restrict__ XWf, const float* __restrict__ dinv,
                const float* __restrict__ bg, float* __restrict__ out, int Nn) {
  __shared__ _Float16 sA[128 * 40];
  __shared__ _Float16 sBc[64 * 40];
  __shared__ _Float16 sBs[64 * 40];
  const int tid = threadIdx.x, lane = tid & 31, wave = tid >> 5;
  const int rowBase = blockIdx.x * 128;
  const int colBase = blockIdx.y * 64;

  v8f z = {0.f, 0.f, 0.f, 0.f, 0.f, 0.f, 0.f, 0.f};
  v8f acc[4] = {z, z, z, z};

  int arow = rowBase + wave * 16 + (lane & 15);   // A-frag M row for this lane
  float cr = (arow < Nn) ? cv[arow] : 0.f;
  float sr = (arow < Nn) ? sv[arow] : 0.f;

  for (int k0 = 0; k0 < DD; k0 += 32) {
    __syncthreads();
#ifdef NWG_ASYNC
    for (int i = tid; i < 512; i += 256) {
      int r = i >> 2, c = (i & 3) * 8;
      int gr = min(rowBase + r, Nn - 1);
      async_cp16(&Qh[(size_t)gr * DD + k0 + c], &sA[r * 40 + c]);
    }
#else
    for (int i = tid; i < 128 * 32; i += 256) {
      int r = i >> 5, kk = i & 31;
      int gr = min(rowBase + r, Nn - 1);
      sA[r * 40 + kk] = Qh[(size_t)gr * DD + k0 + kk];
    }
#endif
    for (int i = tid; i < 32 * 64; i += 256) {
      int kk = i >> 6, nn = i & 63;
      int gidx = (k0 + kk) * DD + colBase + nn;
      sBc[nn * 40 + kk] = KVch[gidx];
      sBs[nn * 40 + kk] = KVsh[gidx];
    }
#ifdef NWG_ASYNC
    NWG_WAIT_ASYNC();
#endif
    __syncthreads();
    int hi = lane >> 4;
    v16h a = load_frag(sA, wave * 16 + (lane & 15), 40, hi);
    v16h a_c = scale_frag(a, cr);
    v16h a_s = scale_frag(a, sr);
#pragma unroll
    for (int t = 0; t < 4; ++t) {
      int n = t * 16 + (lane & 15);
      v16h bc = load_frag(sBc, n, 40, hi);
      acc[t] = __builtin_amdgcn_wmma_f32_16x16x32_f16(false, a_c, false, bc,
                                                      (short)0, acc[t], false, false);
      v16h bs = load_frag(sBs, n, 40, hi);
      acc[t] = __builtin_amdgcn_wmma_f32_16x16x32_f16(false, a_s, false, bs,
                                                      (short)0, acc[t], false, false);
    }
  }

  int hi = lane >> 4;
#pragma unroll
  for (int t = 0; t < 4; ++t) {
    int col = colBase + t * 16 + (lane & 15);
#pragma unroll
    for (int r = 0; r < 8; ++r) {
      int row = rowBase + wave * 16 + hi * 8 + r;
      if (row < Nn) {
        size_t idx = (size_t)row * DD + col;
        float di = dinv[row];
        out[idx] = acc[t][r] / normv[row] + Vf[idx] + bg[col] + di * di * XWf[idx];
      }
    }
  }
}

// ---------------------------------------------------------------------------
// K8: GCN edge scatter: out[dst] += dinv[src]*dinv[dst] * XW[src]  (wave/edge)
// ---------------------------------------------------------------------------
__global__ __launch_bounds__(256)
void nwg_k8_scatter(const int* __restrict__ ei, const float* __restrict__ XWf,
                    const float* __restrict__ dinv, float* out, int E_) {
  int e = blockIdx.x * 8 + (threadIdx.x >> 5);
  if (e >= E_) return;
  int lane = threadIdx.x & 31;
  int src = ei[e], dst = ei[E_ + e];
  float coef = dinv[src] * dinv[dst];
#pragma unroll
  for (int d = lane; d < DD; d += 32)
    atomicAdd(&out[(size_t)dst * DD + d], coef * XWf[(size_t)src * DD + d]);
}

// ---------------------------------------------------------------------------
// K9: final LN(gamma,beta) + relu in place on d_out (one wave per row)
// ---------------------------------------------------------------------------
__global__ __launch_bounds__(256)
void nwg_k9_lnfinal(float* out, const float* gamma, const float* beta, int Nn) {
  int row = blockIdx.x * 8 + (threadIdx.x >> 5);
  int lane = threadIdx.x & 31;
  if (row >= Nn) return;
  size_t base = (size_t)row * DD + lane * 16;
  float v[16]; float s = 0.f;
#pragma unroll
  for (int i = 0; i < 16; ++i) { v[i] = out[base + i]; s += v[i]; }
  float mu = wave_sum(s) * (1.0f / DD);
  float var = 0.f;
#pragma unroll
  for (int i = 0; i < 16; ++i) { float d = v[i] - mu; var += d * d; }
  var = wave_sum(var) * (1.0f / DD);
  float inv = rsqrtf(var + LN_EPS);
#pragma unroll
  for (int i = 0; i < 16; ++i) {
    int d = lane * 16 + i;
    float y = (v[i] - mu) * inv * gamma[d] + beta[d];
    out[base + i] = fmaxf(y, 0.f);
  }
}

// ---------------------------------------------------------------------------
extern "C" void kernel_launch(void* const* d_in, const int* in_sizes, int n_in,
                              void* d_out, int out_size, void* d_ws, size_t ws_size,
                              hipStream_t stream) {
  const float* x   = (const float*)d_in[0];
  const int*   ei  = (const int*)d_in[1];
  const float* Wq  = (const float*)d_in[2];
  const float* bq  = (const float*)d_in[3];
  const float* Wv  = (const float*)d_in[4];
  const float* bv  = (const float*)d_in[5];
  const float* Wg  = (const float*)d_in[6];
  const float* bg  = (const float*)d_in[7];
  const float* gam = (const float*)d_in[8];
  const float* bet = (const float*)d_in[9];
  float* out = (float*)d_out;

  const int Nn = in_sizes[0] / DD;
  const int E_ = in_sizes[1] / 2;
  const size_t ND = (size_t)Nn * DD;

  // carve workspace
  char* p = (char*)d_ws;
  auto carve = [&](size_t bytes) {
    char* q = p; p += (bytes + 255) & ~(size_t)255; return q;
  };
  _Float16* xh   = (_Float16*)carve(ND * 2);
  _Float16* Qh   = (_Float16*)carve(ND * 2);
  _Float16* Vh   = (_Float16*)carve(ND * 2);
  float*    Vf   = (float*)carve(ND * 4);
  float*    XWf  = (float*)carve(ND * 4);
  float*    KVc  = (float*)carve((size_t)DD * DD * 4);
  float*    KVs  = (float*)carve((size_t)DD * DD * 4);
  _Float16* KVch = (_Float16*)carve((size_t)DD * DD * 2);
  _Float16* KVsh = (_Float16*)carve((size_t)DD * DD * 2);
  float*    colc = (float*)carve(DD * 4);
  float*    cols_= (float*)carve(DD * 4);
  float*    H    = (float*)carve((size_t)Nn * 4);
  float*    cv   = (float*)carve((size_t)Nn * 4);
  float*    sv   = (float*)carve((size_t)Nn * 4);
  float*    deg  = (float*)carve((size_t)Nn * 4);
  float*    dinv = (float*)carve((size_t)Nn * 4);
  float*    normv= (float*)carve((size_t)Nn * 4);
  unsigned* hmax = (unsigned*)carve(4);
  unsigned* hmin = (unsigned*)carve(4);
  (void)n_in; (void)out_size; (void)ws_size;

  nwg_k0_init<<<512, 256, 0, stream>>>(KVc, KVs, colc, cols_, deg, hmax, hmin, Nn);
  nwg_kpre_cvt<<<1024, 256, 0, stream>>>(x, xh, (long long)ND);

  dim3 g1((Nn + 127) / 128, DD / 64);
  nwg_k1_gemm3<<<g1, 256, 0, stream>>>(xh, Wq, bq, Wv, bv, Wg, Qh, Vf, Vh, XWf, Nn);

  nwg_k_deg<<<(E_ + 255) / 256, 256, 0, stream>>>(ei, deg, E_);
  nwg_k_dinv<<<(Nn + 255) / 256, 256, 0, stream>>>(deg, dinv, Nn);

  nwg_k2_ln<<<(Nn + 7) / 8, 256, 0, stream>>>(Qh, H, hmax, hmin, Nn);
  nwg_k3_trig<<<(Nn + 255) / 256, 256, 0, stream>>>(H, hmax, hmin, cv, sv, Nn);
  nwg_k4_colsum<<<(Nn + 255) / 256, 512, 0, stream>>>(Qh, cv, sv, colc, cols_, Nn);

  dim3 g5(DD / 64, DD / 64, 25);   // split-K over nodes
  nwg_k5_gram<<<g5, 256, 0, stream>>>(Qh, Vh, cv, sv, KVc, KVs, Nn);
  nwg_k6_cvt<<<(DD * DD + 255) / 256, 256, 0, stream>>>(KVc, KVs, KVch, KVsh);

  nwg_k7b_norm<<<(Nn + 7) / 8, 256, 0, stream>>>(Qh, colc, cols_, cv, sv, normv, Nn);
  dim3 g7((Nn + 127) / 128, DD / 64);
  nwg_k7_out<<<g7, 256, 0, stream>>>(Qh, KVch, KVsh, cv, sv, normv, Vf, XWf,
                                     dinv, bg, out, Nn);

  nwg_k8_scatter<<<(E_ + 7) / 8, 256, 0, stream>>>(ei, XWf, dinv, out, E_);
  nwg_k9_lnfinal<<<(Nn + 7) / 8, 256, 0, stream>>>(out, gam, bet, Nn);
}
